// MyModelWithDWA_35364760716037
// MI455X (gfx1250) — matrix operand order repaired and checked by reading, not compiled, not run on hardware
//
#include <hip/hip_runtime.h>

// ---------------------------------------------------------------------------
// MI455X (gfx1250): wave32, WMMA bf16 16x16x32 (f32 accum).
// Both GEMM operands staged in LDS; B double-buffered in 32-wide K chunks
// filled by GLOBAL_LOAD_ASYNC_TO_LDS_B128 (ASYNCcnt) one chunk ahead.
// ---------------------------------------------------------------------------

#define BB   64
#define TT   512
#define II   256
#define HH   1024
#define FOURH 4096
#define KC   1280          // I + H
#define KCP  1288          // padded LDS row stride for A (LSTM)
#define HP   1032          // padded LDS row stride for A (shared gemm)
#define BSTR 40            // padded LDS row stride for B chunks (32 + 8)
#define FF   11

typedef __bf16 bf16_t;
typedef __attribute__((ext_vector_type(16))) __bf16 v16bf;
typedef __attribute__((ext_vector_type(8)))  float  v8f;

// d_out element offsets (f32): state_out, reward_out, hT, cT
#define STATE_OFF  0
#define REWARD_OFF 360448
#define HT_OFF     393216
#define CT_OFF     458752

__device__ __forceinline__ v8f wmma_bf16(v16bf a, v16bf b, v8f c) {
  return __builtin_amdgcn_wmma_f32_16x16x32_bf16(
      false, a, false, b, (short)0, c, false, false);
}

// Async copy 16B global -> LDS (CDNA5 GLOBAL_LOAD_ASYNC_TO_LDS_B128, ASYNCcnt).
__device__ __forceinline__ void async_g2l_b128(unsigned lds_off,
                                               const void* gptr) {
  asm volatile("global_load_async_to_lds_b128 %0, %1, off"
               :: "v"(lds_off), "v"(gptr) : "memory");
}
__device__ __forceinline__ void wait_async0() {
  asm volatile("s_wait_asynccnt 0" ::: "memory");
}

// A-fragment (16x32 bf16, ISA 7.12.2) from an LDS tile with padded stride.
__device__ __forceinline__ v16bf lds_afrag(const bf16_t* As, int stride,
                                           int l15, int k0, int hl) {
  union { uint4 u[2]; v16bf v; } t;
  const bf16_t* p = As + l15 * stride + k0 + hl * 16;
  t.u[0] = *(const uint4*)(p);
  t.u[1] = *(const uint4*)(p + 8);
  return t.v;
}

// B-fragment from an LDS chunk buffer (row = weight row, 32 K elems + pad).
__device__ __forceinline__ v16bf lds_bfrag(const bf16_t* Bbuf, int row, int hl) {
  union { uint4 u[2]; v16bf v; } t;
  const bf16_t* p = Bbuf + row * BSTR + hl * 16;
  t.u[0] = *(const uint4*)(p);
  t.u[1] = *(const uint4*)(p + 8);
  return t.v;
}

__device__ __forceinline__ float sigm(float x) {
  return 1.0f / (1.0f + __expf(-x));
}

// ---------------------------------------------------------------------------
// Prep kernels
// ---------------------------------------------------------------------------
__global__ void k_cvt_bf16(const float* __restrict__ in,
                           bf16_t* __restrict__ out, int n) {
  for (int i = blockIdx.x * blockDim.x + threadIdx.x; i < n;
       i += gridDim.x * blockDim.x)
    out[i] = (bf16_t)in[i];
}

__global__ void k_build_wcat(const float* __restrict__ Wih,
                             const float* __restrict__ Whh,
                             bf16_t* __restrict__ Wcat) {
  const int n = FOURH * KC;
  for (int i = blockIdx.x * blockDim.x + threadIdx.x; i < n;
       i += gridDim.x * blockDim.x) {
    int r = i / KC, k = i - r * KC;
    float v = (k < II) ? Wih[r * II + k] : Whh[r * HH + (k - II)];
    Wcat[i] = (bf16_t)v;
  }
}

__global__ void k_prep_state(const float* __restrict__ h0,
                             const float* __restrict__ c0,
                             const float* __restrict__ bih,
                             const float* __restrict__ bhh,
                             bf16_t* __restrict__ hprev0,
                             float* __restrict__ hstate,
                             float* __restrict__ cstate,
                             float* __restrict__ bias) {
  int i = blockIdx.x * blockDim.x + threadIdx.x;
  if (i < BB * HH) {
    hprev0[i] = (bf16_t)h0[i];
    hstate[i] = h0[i];
    cstate[i] = c0[i];
  }
  if (i < FOURH) bias[i] = bih[i] + bhh[i];
}

// ---------------------------------------------------------------------------
// LSTM step. 32 blocks x 256 thr (8 waves).
// Block: 16-batch A tile (full K, LDS) x 128 hidden cols x 4 gates.
// B (Wcat rows) streamed through a 2x32-K LDS chunk pipeline via async DMA.
// ---------------------------------------------------------------------------
__global__ __launch_bounds__(256, 1)
void k_lstm_step(const bf16_t* __restrict__ x_bf,
                 const bf16_t* __restrict__ Wcat,
                 const float*  __restrict__ bias,
                 const bf16_t* __restrict__ hprev0,
                 bf16_t* __restrict__ hs,
                 float*  __restrict__ hstate,
                 float*  __restrict__ cstate,
                 int t) {
  __shared__ bf16_t As[16 * KCP];                 // ~41 KB
  __shared__ bf16_t Bs[2][512 * BSTR];            // 2 x 40 KB
  const int tid  = threadIdx.x;
  const int lane = tid & 31;
  const int hl   = lane >> 4;
  const int l15  = lane & 15;
  const int wv   = tid >> 5;                      // wave in block: 0..7
  const int m0    = (blockIdx.x >> 3) << 4;       // batch row tile (block)
  const int nbase = (blockIdx.x & 7) * 128;       // hidden col base (block)

  const unsigned abase  = (unsigned)(unsigned long long)(void*)As;
  const unsigned bbase0 = (unsigned)(unsigned long long)(void*)&Bs[0][0];
  const unsigned bbase1 = (unsigned)(unsigned long long)(void*)&Bs[1][0];

  // ---- issue async fill: A tile [16 x KC] = [x_t | h_{t-1}] ----
  for (int c = tid; c < 16 * (KC / 8); c += 256) {     // 2560 x b128
    const int row  = c / (KC / 8);
    const int col8 = (c - row * (KC / 8)) * 8;
    const int rb   = m0 + row;
    const bf16_t* g;
    if (col8 < II) {
      g = x_bf + (size_t)(rb * TT + t) * II + col8;
    } else {
      const bf16_t* hrow = (t == 0)
          ? (hprev0 + (size_t)rb * HH)
          : (hs + (size_t)(rb * TT + (t - 1)) * HH);
      g = hrow + (col8 - II);
    }
    async_g2l_b128(abase + (unsigned)(row * KCP + col8) * 2u, g);
  }
  // ---- issue async fill: B chunk 0 (512 rows x 32 K) ----
  for (int c = tid; c < 2048; c += 256) {              // 4 x b128 per row
    const int row = c >> 2;
    const int k8  = (c & 3) * 8;
    const int gate = row >> 7, ln = row & 127;
    const bf16_t* g = Wcat + (size_t)(gate * HH + nbase + ln) * KC + k8;
    async_g2l_b128(bbase0 + (unsigned)(row * BSTR + k8) * 2u, g);
  }
  wait_async0();
  __syncthreads();

  v8f zi = {}, zf = {}, zg = {}, zo = {};
  const int r0 = 0 * 128 + wv * 16 + l15;         // gate-i LDS row
  const int r1 = 1 * 128 + wv * 16 + l15;
  const int r2 = 2 * 128 + wv * 16 + l15;
  const int r3 = 3 * 128 + wv * 16 + l15;

  const int NCH = KC / 32;                        // 40 chunks
  for (int ch = 0; ch < NCH; ++ch) {
    const int cur = ch & 1;
    // prefetch next chunk into the other buffer
    if (ch + 1 < NCH) {
      const int kc = (ch + 1) * 32;
      const unsigned nb = (cur ? bbase0 : bbase1);
      for (int c = tid; c < 2048; c += 256) {
        const int row = c >> 2;
        const int k8  = (c & 3) * 8;
        const int gate = row >> 7, ln = row & 127;
        const bf16_t* g = Wcat + (size_t)(gate * HH + nbase + ln) * KC + kc + k8;
        async_g2l_b128(nb + (unsigned)(row * BSTR + k8) * 2u, g);
      }
    }
    // compute current chunk from LDS only
    const bf16_t* bb = &Bs[cur][0];
    const v16bf a = lds_afrag(As, KCP, l15, ch * 32, hl);
    zi = wmma_bf16(a, lds_bfrag(bb, r0, hl), zi);
    zf = wmma_bf16(a, lds_bfrag(bb, r1, hl), zf);
    zg = wmma_bf16(a, lds_bfrag(bb, r2, hl), zg);
    zo = wmma_bf16(a, lds_bfrag(bb, r3, hl), zo);

    wait_async0();
    __syncthreads();
  }

  const int ncol = nbase + wv * 16 + l15;
  const float bi  = bias[0 * HH + ncol];
  const float bfv = bias[1 * HH + ncol];
  const float bg  = bias[2 * HH + ncol];
  const float bo  = bias[3 * HH + ncol];

#pragma unroll
  for (int r = 0; r < 8; ++r) {
    const int bidx = m0 + r + hl * 8;             // batch index
    const float iv = sigm(zi[r] + bi);
    const float fv = sigm(zf[r] + bfv);
    const float gv = tanhf(zg[r] + bg);
    const float ov = sigm(zo[r] + bo);
    const size_t ci = (size_t)bidx * HH + ncol;
    const float cn = fv * cstate[ci] + iv * gv;
    const float hn = ov * tanhf(cn);
    cstate[ci] = cn;
    hstate[ci] = hn;
    hs[(size_t)(bidx * TT + t) * HH + ncol] = (bf16_t)hn;
  }
}

// ---------------------------------------------------------------------------
// shared = hs @ W_sh^T + b_sh  (M=32768, N=1024, K=1024), bf16 out.
// Block: 16-row A tile (full K, LDS) x 512 output cols; B chunk pipeline.
// ---------------------------------------------------------------------------
__global__ __launch_bounds__(256, 1)
void k_shared_gemm(const bf16_t* __restrict__ hs,
                   const bf16_t* __restrict__ Wsh,
                   const float*  __restrict__ b_sh,
                   bf16_t* __restrict__ sh_out) {
  __shared__ bf16_t As[16 * HP];                  // ~33 KB
  __shared__ bf16_t Bs[2][512 * BSTR];            // 2 x 40 KB
  const int tid  = threadIdx.x;
  const int lane = tid & 31;
  const int hl   = lane >> 4;
  const int l15  = lane & 15;
  const int wv   = tid >> 5;                      // wave in block: 0..7
  const int m0    = ((blockIdx.x * 8) >> 4) << 4; // per-block row tile
  const int nbase = (blockIdx.x & 1) * 512;       // per-block col base

  const unsigned abase  = (unsigned)(unsigned long long)(void*)As;
  const unsigned bbase0 = (unsigned)(unsigned long long)(void*)&Bs[0][0];
  const unsigned bbase1 = (unsigned)(unsigned long long)(void*)&Bs[1][0];

  // ---- issue async fill: A tile [16 x 1024] ----
  for (int c = tid; c < 16 * (HH / 8); c += 256) {     // 2048 x b128
    const int row  = c >> 7;
    const int col8 = (c & 127) * 8;
    const bf16_t* g = hs + (size_t)(m0 + row) * HH + col8;
    async_g2l_b128(abase + (unsigned)(row * HP + col8) * 2u, g);
  }
  // ---- issue async fill: B chunk 0 ----
  for (int c = tid; c < 2048; c += 256) {
    const int row = c >> 2;
    const int k8  = (c & 3) * 8;
    const bf16_t* g = Wsh + (size_t)(nbase + row) * HH + k8;
    async_g2l_b128(bbase0 + (unsigned)(row * BSTR + k8) * 2u, g);
  }
  wait_async0();
  __syncthreads();

  v8f a0 = {}, a1 = {}, a2 = {}, a3 = {};
  const int r0 = wv * 64 + 0 * 16 + l15;
  const int r1 = wv * 64 + 1 * 16 + l15;
  const int r2 = wv * 64 + 2 * 16 + l15;
  const int r3 = wv * 64 + 3 * 16 + l15;

  const int NCH = HH / 32;                        // 32 chunks
  for (int ch = 0; ch < NCH; ++ch) {
    const int cur = ch & 1;
    if (ch + 1 < NCH) {
      const int kc = (ch + 1) * 32;
      const unsigned nb = (cur ? bbase0 : bbase1);
      for (int c = tid; c < 2048; c += 256) {
        const int row = c >> 2;
        const int k8  = (c & 3) * 8;
        const bf16_t* g = Wsh + (size_t)(nbase + row) * HH + kc + k8;
        async_g2l_b128(nb + (unsigned)(row * BSTR + k8) * 2u, g);
      }
    }
    const bf16_t* bb = &Bs[cur][0];
    const v16bf a = lds_afrag(As, HP, l15, ch * 32, hl);
    a0 = wmma_bf16(a, lds_bfrag(bb, r0, hl), a0);
    a1 = wmma_bf16(a, lds_bfrag(bb, r1, hl), a1);
    a2 = wmma_bf16(a, lds_bfrag(bb, r2, hl), a2);
    a3 = wmma_bf16(a, lds_bfrag(bb, r3, hl), a3);

    wait_async0();
    __syncthreads();
  }

#pragma unroll
  for (int q = 0; q < 4; ++q) {
    const v8f acc = (q == 0) ? a0 : (q == 1) ? a1 : (q == 2) ? a2 : a3;
    const int n = nbase + wv * 64 + q * 16 + l15;
    const float bn = b_sh[n];
#pragma unroll
    for (int r = 0; r < 8; ++r) {
      const int m = m0 + r + hl * 8;
      sh_out[(size_t)m * HH + n] = (bf16_t)(acc[r] + bn);
    }
  }
}

// ---------------------------------------------------------------------------
// Fused causal attention + heads via online softmax on head-projected
// values (heads are linear => carry (m, s, W@acc) only). One wave per batch.
// ---------------------------------------------------------------------------
__global__ __launch_bounds__(32)
void k_attn_heads(const bf16_t* __restrict__ sh,
                  const float* __restrict__ w_att_s, const float* __restrict__ b_att_s,
                  const float* __restrict__ w_att_r, const float* __restrict__ b_att_r,
                  const float* __restrict__ W_state, const float* __restrict__ b_state,
                  const float* __restrict__ W_reward, const float* __restrict__ b_reward,
                  float* __restrict__ state_out, float* __restrict__ reward_out) {
  __shared__ float V[14 * HH];
  const int lane = threadIdx.x;
  const int b = blockIdx.x;

  for (int idx = lane; idx < 14 * HH; idx += 32) {
    const int i = idx >> 10, h = idx & (HH - 1);
    float v;
    if (i == 0)      v = w_att_s[h];
    else if (i == 1) v = w_att_r[h];
    else if (i < 13) v = W_state[(i - 2) * HH + h];
    else             v = W_reward[h];
    V[idx] = v;
  }
  __syncthreads();

  const float bas = b_att_s[0], bar = b_att_r[0], brw = b_reward[0];
  float bst[FF];
#pragma unroll
  for (int f = 0; f < FF; ++f) bst[f] = b_state[f];

  float ms = -__builtin_inff(), ss = 0.f;
  float mr = -__builtin_inff(), sr = 0.f;
  float wsacc[FF];
#pragma unroll
  for (int f = 0; f < FF; ++f) wsacc[f] = 0.f;
  float wracc = 0.f;

  for (int t = 0; t < TT; ++t) {
    const bf16_t* frow = sh + (size_t)(b * TT + t) * HH;
    float fv[32];
#pragma unroll
    for (int j = 0; j < 32; ++j) fv[j] = (float)frow[j * 32 + lane];

    float p[14];
#pragma unroll
    for (int i = 0; i < 14; ++i) {
      float s = 0.f;
#pragma unroll
      for (int j = 0; j < 32; ++j) s += V[i * HH + j * 32 + lane] * fv[j];
#pragma unroll
      for (int off = 16; off >= 1; off >>= 1) s += __shfl_xor(s, off, 32);
      p[i] = s;
    }

    const float ls = p[0] + bas;
    const float lr = p[1] + bar;

    float mn = fmaxf(ms, ls);
    float sc = __expf(ms - mn);
    float e  = __expf(ls - mn);
    ss = ss * sc + e;
#pragma unroll
    for (int f = 0; f < FF; ++f) wsacc[f] = wsacc[f] * sc + e * p[2 + f];
    ms = mn;

    mn = fmaxf(mr, lr);
    sc = __expf(mr - mn);
    e  = __expf(lr - mn);
    sr = sr * sc + e;
    wracc = wracc * sc + e * p[13];
    mr = mn;

    if (lane == 0) {
      const size_t o = (size_t)(b * TT + t);
#pragma unroll
      for (int f = 0; f < FF; ++f)
        state_out[o * FF + f] = wsacc[f] / ss + bst[f];
      reward_out[o] = wracc / sr + brw;
    }
  }
}

// ---------------------------------------------------------------------------
// Host launcher
// ---------------------------------------------------------------------------
extern "C" void kernel_launch(void* const* d_in, const int* in_sizes, int n_in,
                              void* d_out, int out_size, void* d_ws, size_t ws_size,
                              hipStream_t stream) {
  (void)in_sizes; (void)n_in; (void)out_size; (void)ws_size;

  const float* x    = (const float*)d_in[0];
  const float* h0   = (const float*)d_in[2];
  const float* c0   = (const float*)d_in[3];
  const float* Wih  = (const float*)d_in[4];
  const float* Whh  = (const float*)d_in[5];
  const float* bih  = (const float*)d_in[6];
  const float* bhh  = (const float*)d_in[7];
  const float* Wsh  = (const float*)d_in[8];
  const float* bsh  = (const float*)d_in[9];
  const float* was  = (const float*)d_in[10];
  const float* bas  = (const float*)d_in[11];
  const float* war  = (const float*)d_in[12];
  const float* barr = (const float*)d_in[13];
  const float* Wst  = (const float*)d_in[14];
  const float* bst  = (const float*)d_in[15];
  const float* Wrw  = (const float*)d_in[16];
  const float* brw  = (const float*)d_in[17];

  float* out = (float*)d_out;
  float* hstate = out + HT_OFF;
  float* cstate = out + CT_OFF;

  char* ws = (char*)d_ws;
  size_t o = 0;
  bf16_t* x_bf   = (bf16_t*)(ws + o); o += (size_t)BB * TT * II * 2;     // 16 MB
  bf16_t* Wcat   = (bf16_t*)(ws + o); o += (size_t)FOURH * KC * 2;       // 10 MB
  bf16_t* Wsh_bf = (bf16_t*)(ws + o); o += (size_t)HH * HH * 2;          //  2 MB
  float*  bias   = (float*) (ws + o); o += (size_t)FOURH * 4;
  bf16_t* hprev0 = (bf16_t*)(ws + o); o += (size_t)BB * HH * 2;
  bf16_t* hs     = (bf16_t*)(ws + o); o += (size_t)BB * TT * HH * 2;     // 64 MB
  bf16_t* sh_bf  = (bf16_t*)(ws + o); o += (size_t)BB * TT * HH * 2;     // 64 MB

  // prep
  k_cvt_bf16<<<4096, 256, 0, stream>>>(x, x_bf, BB * TT * II);
  k_cvt_bf16<<<2048, 256, 0, stream>>>(Wsh, Wsh_bf, HH * HH);
  k_build_wcat<<<4096, 256, 0, stream>>>(Wih, Whh, Wcat);
  k_prep_state<<<(BB * HH + 255) / 256, 256, 0, stream>>>(
      h0, c0, bih, bhh, hprev0, hstate, cstate, bias);

  // sequential LSTM: 512 graph-captured step launches
  for (int t = 0; t < TT; ++t) {
    k_lstm_step<<<32, 256, 0, stream>>>(x_bf, Wcat, bias, hprev0,
                                        hs, hstate, cstate, t);
  }

  // shared projection
  k_shared_gemm<<<4096, 256, 0, stream>>>(hs, Wsh_bf, bsh, sh_bf);

  // fused causal attention + output heads
  k_attn_heads<<<BB, 32, 0, stream>>>(sh_bf, was, bas, war, barr,
                                      Wst, bst, Wrw, brw,
                                      out + STATE_OFF, out + REWARD_OFF);
}
